// ResidualVQ_4200478015564
// MI455X (gfx1250) — matrix operand order repaired and checked by reading, not compiled, not run on hardware
//
#include <hip/hip_runtime.h>
#include <stdint.h>

// ---------------------------------------------------------------------------
// ResidualVQ forward for MI455X (gfx1250, wave32, WMMA, async LDS copy).
// Per quantizer step: scores = ||e||^2 - 2*(r . e) via v_wmma_f32_16x16x32_bf16
// (bf16 codebook pre-converted once), argmin over C=1024, fused gather/update.
// Double-buffered LDS staging of 16-code tiles via GLOBAL_LOAD_ASYNC_TO_LDS.
// ---------------------------------------------------------------------------

typedef __attribute__((ext_vector_type(16))) __bf16 v16bf;
typedef __attribute__((ext_vector_type(8)))  float  v8f;
typedef int v4i __attribute__((vector_size(16)));    // matches builtin param type

#define AS1 __attribute__((address_space(1)))
#define AS3 __attribute__((address_space(3)))

#if __has_builtin(__builtin_amdgcn_global_load_async_to_lds_b128)
#define USE_ASYNC_LDS 1
#else
#define USE_ASYNC_LDS 0
#endif

union BFrag {
    v16bf    v;
    uint32_t u[8];
    uint4    q[2];
};

constexpr int Dv   = 512;
constexpr int Cv   = 1024;
constexpr int Qv   = 8;
constexpr int Tv   = 32768;       // B*N tokens
constexpr int BSTR = 520;         // bf16 elems per LDS code row (16B-aligned, bank-spread)
constexpr int TOKWG = 128;        // tokens per workgroup
constexpr int NWG  = Tv / TOKWG;  // 256 workgroups per step
constexpr int NCH  = Cv / 16;     // 64 code chunks

// round-to-nearest-even f32 -> bf16, packed pair into one dword (lo in [15:0])
__device__ __forceinline__ uint32_t pk_bf16(float lo, float hi) {
    union { float f; uint32_t u; } a{lo}, b{hi};
    uint32_t ra = a.u + 0x7FFFu + ((a.u >> 16) & 1u);
    uint32_t rb = b.u + 0x7FFFu + ((b.u >> 16) & 1u);
    return (ra >> 16) | (rb & 0xFFFF0000u);
}

// ---------------------------------------------------------------------------
// Prep: per-code squared norms (f32) + one-time f32->bf16 codebook conversion.
// One wave per code (Q*C blocks).
// ---------------------------------------------------------------------------
__global__ __launch_bounds__(32)
void vq_prep_kernel(const float* __restrict__ cb,
                    unsigned short* __restrict__ cbh,
                    float* __restrict__ norms) {
    const int c  = blockIdx.x;                   // 0 .. Q*C-1
    const int ln = threadIdx.x;
    const float4* p = (const float4*)(cb + (size_t)c * Dv);
    float s = 0.0f;
    uint32_t pkd[8];
    #pragma unroll
    for (int j = 0; j < 4; ++j) {                // lane owns 16 contiguous floats
        float4 f = p[ln * 4 + j];
        s += f.x * f.x + f.y * f.y + f.z * f.z + f.w * f.w;
        pkd[2 * j]     = pk_bf16(f.x, f.y);
        pkd[2 * j + 1] = pk_bf16(f.z, f.w);
    }
    uint4* dst = (uint4*)(cbh + (size_t)c * Dv + ln * 16);
    dst[0] = make_uint4(pkd[0], pkd[1], pkd[2], pkd[3]);
    dst[1] = make_uint4(pkd[4], pkd[5], pkd[6], pkd[7]);
    #pragma unroll
    for (int off = 16; off >= 1; off >>= 1) s += __shfl_xor(s, off, 32);
    if (ln == 0) norms[c] = s;
}

// ---------------------------------------------------------------------------
// Staging helpers: 64B per thread (4 x b128), global bf16 -> LDS.
// ---------------------------------------------------------------------------
__device__ __forceinline__ void stage_issue(const unsigned short* src,
                                            unsigned short* dst) {
#if USE_ASYNC_LDS
    #pragma unroll
    for (int j = 0; j < 4; ++j)
        __builtin_amdgcn_global_load_async_to_lds_b128(
            (AS1 v4i*)(src + 8 * j), (AS3 v4i*)(dst + 8 * j), 0, 0);
#else
    uint4 t0 = ((const uint4*)src)[0], t1 = ((const uint4*)src)[1];
    uint4 t2 = ((const uint4*)src)[2], t3 = ((const uint4*)src)[3];
    ((uint4*)dst)[0] = t0; ((uint4*)dst)[1] = t1;
    ((uint4*)dst)[2] = t2; ((uint4*)dst)[3] = t3;
#endif
}

__device__ __forceinline__ void wait_async() {
#if USE_ASYNC_LDS
#if __has_builtin(__builtin_amdgcn_s_wait_asynccnt)
    __builtin_amdgcn_s_wait_asynccnt(0);
#else
    asm volatile("s_wait_asynccnt 0x0" ::: "memory");
#endif
#endif
}

// ---------------------------------------------------------------------------
// One quantizer step. 256 threads = 8 waves; WG owns 128 tokens, wave owns 16.
// ---------------------------------------------------------------------------
__global__ __launch_bounds__(256)
void vq_step_kernel(const float* __restrict__ cb,           // [C,512] f32 (this quantizer)
                    const unsigned short* __restrict__ cbh, // [C,512] bf16 (this quantizer)
                    const float* __restrict__ norms,        // [C] f32
                    float* __restrict__ residual,           // [T,512] f32 (ws)
                    float* __restrict__ qout,               // [T,512] f32 (d_out)
                    float* __restrict__ out_idx,            // [T,Q] as f32 (d_out)
                    float* __restrict__ commit_part,        // [NWG] this quantizer's slice
                    int qstep, int first) {
    __shared__ unsigned short lds_b[2 * 16 * BSTR];  // double-buffered 16-code tile
    __shared__ int   lds_idx[TOKWG];
    __shared__ float lds_cs[8];

    const int tid  = threadIdx.x;
    const int lane = tid & 31;
    const int wave = tid >> 5;                   // 0..7
    const int col  = lane & 15;
    const int hlf  = lane >> 4;                  // K-half select (contiguous k order)
    const int tok_base = blockIdx.x * TOKWG;
    const int wtok = tok_base + wave * 16;

    // ---- A fragments: 16 tokens x 512, f32 -> bf16, in contiguous k order (matches B).
    // lane holds row=col; elems 0..15 <- K=k0+hlf*16.. (contiguous, matches B)
    const float* rp = residual + (size_t)(wtok + col) * Dv;
    v16bf afrag[16];
    #pragma unroll
    for (int kf = 0; kf < 16; ++kf) {
        const int k0 = kf * 32;
        const float4* p0 = (const float4*)(rp + k0 + hlf * 16);
        const float4* p1 = (const float4*)(rp + k0 + hlf * 16 + 8);
        float4 f0 = p0[0], f1 = p0[1];
        float4 f2 = p1[0], f3 = p1[1];
        BFrag af;
        af.u[0] = pk_bf16(f0.x, f0.y); af.u[1] = pk_bf16(f0.z, f0.w);
        af.u[2] = pk_bf16(f1.x, f1.y); af.u[3] = pk_bf16(f1.z, f1.w);
        af.u[4] = pk_bf16(f2.x, f2.y); af.u[5] = pk_bf16(f2.z, f2.w);
        af.u[6] = pk_bf16(f3.x, f3.y); af.u[7] = pk_bf16(f3.z, f3.w);
        afrag[kf] = af.v;
    }

    float rmin[8];
    int   ridx[8];
    #pragma unroll
    for (int m = 0; m < 8; ++m) { rmin[m] = 3.0e38f; ridx[m] = 0; }

    // staging coords: thread copies 32 bf16 of one code row
    const int crow = tid >> 4;                   // code row 0..15
    const int ccol = (tid & 15) * 32;            // col base

    // prologue: issue chunk 0 into buffer 0
    stage_issue(cbh + (size_t)crow * Dv + ccol, lds_b + crow * BSTR + ccol);

    for (int ch = 0; ch < NCH; ++ch) {
        wait_async();                             // my async writes for buf[ch&1] done
        __syncthreads();                          // everyone's writes done / prev reads done

        // issue next chunk into the other buffer (overlaps with WMMA below)
        if (ch + 1 < NCH) {
            stage_issue(cbh + (size_t)((ch + 1) * 16 + crow) * Dv + ccol,
                        lds_b + ((ch + 1) & 1) * (16 * BSTR) + crow * BSTR + ccol);
        }

        // ---- 16 WMMAs over K=512, two accumulators for ILP
        v8f acc0 = {}; v8f acc1 = {};
        const unsigned short* bp = lds_b + (ch & 1) * (16 * BSTR) + col * BSTR + hlf * 16;
        #pragma unroll
        for (int kf = 0; kf < 16; kf += 2) {
            BFrag b0, b1;
            b0.q[0] = ((const uint4*)(bp + kf * 32))[0];
            b0.q[1] = ((const uint4*)(bp + kf * 32))[1];
            b1.q[0] = ((const uint4*)(bp + (kf + 1) * 32))[0];
            b1.q[1] = ((const uint4*)(bp + (kf + 1) * 32))[1];
            acc0 = __builtin_amdgcn_wmma_f32_16x16x32_bf16(
                       false, afrag[kf],     false, b0.v, (short)0, acc0, false, false);
            acc1 = __builtin_amdgcn_wmma_f32_16x16x32_bf16(
                       false, afrag[kf + 1], false, b1.v, (short)0, acc1, false, false);
        }

        // ---- running argmin (||r||^2 constant per token -> omitted)
        const int   code = ch * 16 + col;
        const float nrm  = norms[code];
        #pragma unroll
        for (int m = 0; m < 8; ++m) {
            float s = nrm - 2.0f * (acc0[m] + acc1[m]);
            if (s < rmin[m]) { rmin[m] = s; ridx[m] = code; }
        }
    }

    // ---- cross-lane argmin within each 16-lane half (rows m / m+8), lowest-idx ties
    #pragma unroll
    for (int m = 0; m < 8; ++m) {
        float v = rmin[m]; int ix = ridx[m];
        #pragma unroll
        for (int off = 8; off >= 1; off >>= 1) {
            float ov = __shfl_xor(v, off, 32);
            int   oi = __shfl_xor(ix, off, 32);
            if (ov < v || (ov == v && oi < ix)) { v = ov; ix = oi; }
        }
        if (col == 0) lds_idx[wave * 16 + hlf * 8 + m] = ix;
    }
    __syncthreads();

    // ---- fused update: q = embed[idx] (f32); qout += q; residual -= q; commit += (q-r)^2
    float csum = 0.0f;
    {
        const int tl = tid >> 1;                  // token local 0..127
        const int d0 = (tid & 1) * 256;
        const int tg = tok_base + tl;
        const int code = lds_idx[tl];
        const float4* e  = (const float4*)(cb + (size_t)code * Dv + d0);
        float4* r  = (float4*)(residual + (size_t)tg * Dv + d0);
        float4* qo = (float4*)(qout + (size_t)tg * Dv + d0);
        #pragma unroll 4
        for (int j = 0; j < 64; ++j) {            // 64 x float4 = 256 floats
            float4 ev = e[j];
            float4 rv = r[j];
            float dx = ev.x - rv.x, dy = ev.y - rv.y, dz = ev.z - rv.z, dw = ev.w - rv.w;
            csum += dx * dx + dy * dy + dz * dz + dw * dw;
            float4 qv;
            if (first) qv = ev;
            else { float4 o = qo[j]; qv = make_float4(o.x + ev.x, o.y + ev.y, o.z + ev.z, o.w + ev.w); }
            qo[j] = qv;
            r[j]  = make_float4(rv.x - ev.x, rv.y - ev.y, rv.z - ev.z, rv.w - ev.w);
        }
        if ((tid & 1) == 0)
            out_idx[(size_t)tg * Qv + qstep] = (float)code;
    }

    // deterministic commit reduction: wave shuffle -> LDS -> single ordered sum
    #pragma unroll
    for (int off = 16; off >= 1; off >>= 1) csum += __shfl_xor(csum, off, 32);
    if (lane == 0) lds_cs[wave] = csum;
    __syncthreads();
    if (tid == 0) {
        float t = 0.0f;
        #pragma unroll
        for (int i = 0; i < 8; ++i) t += lds_cs[i];
        commit_part[blockIdx.x] = t;
    }
}

// ---------------------------------------------------------------------------
// Finalize: commits[q] = (ordered sum of WG partials) / (B*N*D); mean over Q.
// ---------------------------------------------------------------------------
__global__ __launch_bounds__(32)
void vq_finalize_kernel(const float* __restrict__ commit_part,
                        float* __restrict__ out_tail) {
    if (threadIdx.x == 0) {
        const float invBND = 1.0f / 16777216.0f;  // 8*4096*512
        float msum = 0.0f;
        for (int qd = 0; qd < Qv; ++qd) {
            float s = 0.0f;
            for (int w = 0; w < NWG; ++w) s += commit_part[qd * NWG + w];
            float ci = s * invBND;
            out_tail[qd] = ci;
            msum += ci;
        }
        out_tail[8] = msum * 0.125f;
    }
}

// ---------------------------------------------------------------------------
extern "C" void kernel_launch(void* const* d_in, const int* in_sizes, int n_in,
                              void* d_out, int out_size, void* d_ws, size_t ws_size,
                              hipStream_t stream) {
    const float* x  = (const float*)d_in[0];     // [8,4096,512]
    const float* cb = (const float*)d_in[1];     // [8,1024,512]

    float* qout     = (float*)d_out;             // T*D
    float* out_idx  = qout + (size_t)Tv * Dv;    // T*Q (indices as f32)
    float* out_tail = out_idx + (size_t)Tv * Qv; // 8 commits + mean

    float*          residual    = (float*)d_ws;                    // T*D f32 (64 MB)
    unsigned short* cbh         = (unsigned short*)(residual + (size_t)Tv * Dv); // Q*C*D bf16 (8 MB)
    float*          norms       = (float*)(cbh + (size_t)Qv * Cv * Dv);          // Q*C
    float*          commit_part = norms + (size_t)Qv * Cv;                       // Q*NWG

    (void)hipMemcpyAsync(residual, x, (size_t)Tv * Dv * sizeof(float),
                         hipMemcpyDeviceToDevice, stream);

    vq_prep_kernel<<<Qv * Cv, 32, 0, stream>>>(cb, cbh, norms);

    for (int q = 0; q < Qv; ++q) {
        vq_step_kernel<<<NWG, 256, 0, stream>>>(
            cb + (size_t)q * Cv * Dv,
            cbh + (size_t)q * Cv * Dv,
            norms + (size_t)q * Cv,
            residual, qout, out_idx,
            commit_part + (size_t)q * NWG,
            q, (q == 0) ? 1 : 0);
    }

    vq_finalize_kernel<<<1, 32, 0, stream>>>(commit_part, out_tail);
}
